// ResidualAttention_65558380806657
// MI455X (gfx1250) — matrix operand order repaired and checked by reading, not compiled, not run on hardware
//
#include <hip/hip_runtime.h>
#include <hip/hip_bf16.h>

typedef __attribute__((ext_vector_type(16))) _Float16 v16h;
typedef __attribute__((ext_vector_type(8)))  _Float16 v8h;
typedef __attribute__((ext_vector_type(8)))  float    v8f;

#define NB     8
#define C_IN   256
#define C_OUT  256
#define HH     128
#define WWID   128
#define WDIM   512
#define CI_CHUNK 32
#define ROWP   136                      // padded row stride in halfs (mult of 8 -> 16B aligned)
#define CPS    (CI_CHUNK * 3 * ROWP)    // halfs per shifted copy (13056)

// ---------------- K0: styles[n][ci] = (w[n,:] . affine_w[ci,:]) / sqrt(512) + affine_b[ci]
__global__ void k_styles(const float* __restrict__ w, const float* __restrict__ aw,
                         const float* __restrict__ ab, float* __restrict__ styles) {
    const int n  = blockIdx.x;
    const int ci = threadIdx.x;
    const float* wr = w  + (size_t)n  * WDIM;
    const float* ar = aw + (size_t)ci * WDIM;
    float s = 0.f;
    for (int j = 0; j < WDIM; ++j) s += wr[j] * ar[j];
    styles[n * C_IN + ci] = s * 0.044194173824159216f + ab[ci];
}

// ---------------- K1: demodulated per-sample weights, f16, layout [n][tap][co][ci]
__global__ void k_modw(const float* __restrict__ weight, const float* __restrict__ styles,
                       _Float16* __restrict__ wwt) {
    const int co = blockIdx.x;
    const int n  = blockIdx.y;
    const int ci = threadIdx.x;
    const float s = styles[n * C_IN + ci];
    const float* wp = weight + ((size_t)co * C_IN + ci) * 9;
    float wv[9];
    float acc = 0.f;
#pragma unroll
    for (int t = 0; t < 9; ++t) { wv[t] = wp[t] * s; acc += wv[t] * wv[t]; }
    __shared__ float red[256];
    red[ci] = acc;
    __syncthreads();
    for (int st = 128; st > 0; st >>= 1) {
        if (ci < st) red[ci] += red[ci + st];
        __syncthreads();
    }
    const float d = rsqrtf(red[0] + 1e-8f);
#pragma unroll
    for (int t = 0; t < 9; ++t)
        wwt[(((size_t)n * 9 + t) * C_OUT + co) * C_IN + ci] = (_Float16)(wv[t] * d);
}

__device__ __forceinline__ v16h cat8(v8h a, v8h b) {
    return __builtin_shufflevector(a, b, 0,1,2,3,4,5,6,7,8,9,10,11,12,13,14,15);
}

// ---------------- K2: implicit-GEMM 3x3 modulated conv + lrelu*sqrt2 + sigmoid
// grid: (4 co-tiles, 128 rows, 8 samples), block: 256 (8 waves)
// LDS holds three kx-shifted f16 copies of the x tile ; each B fragment is
// 16 strided 2-byte LDS loads (one per ci row).  copy_s[i] = x[i + s - 1].
__global__ __launch_bounds__(256)
void k_conv(const float* __restrict__ x, const _Float16* __restrict__ wwt,
            const float* __restrict__ bias, float* __restrict__ out) {
    const int coTile = blockIdx.x;   // 0..3  -> 64 co each
    const int h      = blockIdx.y;   // 0..127
    const int n      = blockIdx.z;   // 0..7
    const int tid    = threadIdx.x;
    const int lane   = tid & 31;
    const int wave   = tid >> 5;
    const int coSub  = wave & 3;     // which 16-co strip
    const int pxHalf = wave >> 2;    // 0: px 0..63, 1: px 64..127
    const int coBase = coTile * 64 + coSub * 16;

    __shared__ _Float16 tile[3 * CPS];   // ~76.5 KB

    v8f acc[4] = {};
    const int m  = lane & 15;            // A row within 16-co strip
    const int kb = (lane >> 4) << 3;     // 0 or 8 (A K sub-block)
    const _Float16* wbase = wwt + (size_t)n * 9 * C_OUT * C_IN;

    for (int c = 0; c < C_IN / CI_CHUNK; ++c) {
        const int ci0 = c * CI_CHUNK;
        __syncthreads();   // previous chunk's compute done before overwrite

        // halo zeros actually read: copy0[row][0] (x[-1]) and copy2[row][127] (x[128]).
        // Disjoint from stored ranges -> no extra barrier needed.
        if (tid < 96) {
            tile[0 * CPS + tid * ROWP + 0]   = (_Float16)0.f;
            tile[2 * CPS + tid * ROWP + 127] = (_Float16)0.f;
        }
        // stage x[n][ci0..+31][h-1..h+1][:] as f16 into 3 shifted copies.
        // 96 rows x 128 px = 3072 float4 loads; 12 per thread.
        for (int i = 0; i < 12; ++i) {
            const int idx = i * 256 + tid;       // 0..3071
            const int row = idx >> 5;            // 0..95
            const int px  = (idx & 31) << 2;     // 0,4,...,124
            const int ci  = row / 3;
            const int ry  = row - ci * 3;
            const int gh  = h + ry - 1;
            float4 v = make_float4(0.f, 0.f, 0.f, 0.f);
            if ((unsigned)gh < (unsigned)HH)
                v = *(const float4*)&x[(((size_t)n * C_IN + ci0 + ci) * HH + gh) * WWID + px];
            _Float16 hv[4];
            hv[0] = (_Float16)v.x; hv[1] = (_Float16)v.y;
            hv[2] = (_Float16)v.z; hv[3] = (_Float16)v.w;
            const int rb = (ci * 3 + ry) * ROWP;
#pragma unroll
            for (int q = 0; q < 4; ++q) {
                tile[0 * CPS + rb + px + 1 + q] = hv[q];       // copy0: idx = px+1
                tile[1 * CPS + rb + px + q]     = hv[q];       // copy1: idx = px
            }
#pragma unroll
            for (int q = 0; q < 4; ++q) {                      // copy2: idx = px-1
                const int i2 = px - 1 + q;
                if (i2 >= 0) tile[2 * CPS + rb + i2] = hv[q];
            }
        }
        __syncthreads();

#pragma unroll
        for (int ky = 0; ky < 3; ++ky) {
            // A fragments for the 3 taps of this ky (row-major [co][ci] tile)
            v16h A[3];
#pragma unroll
            for (int kx = 0; kx < 3; ++kx) {
                const _Float16* ap = wbase +
                    ((size_t)(ky * 3 + kx) * C_OUT + coBase + m) * C_IN + ci0 + kb;
                v8h alo = *(const v8h*)ap;          // K = kb..kb+7
                v8h ahi = *(const v8h*)(ap + 16);   // K = 16+kb..16+kb+7
                A[kx] = cat8(alo, ahi);
            }
#pragma unroll
            for (int j = 0; j < 4; ++j) {
                const int px0 = pxHalf * 64 + j * 16;
#pragma unroll
                for (int s = 0; s < 3; ++s) {
                    const _Float16* lp = tile + s * CPS + ky * ROWP + px0 + (lane & 15);
                    v16h B;
#pragma unroll
                    for (int e = 0; e < 16; ++e) {
                        const int kk = (e < 8) ? (kb + e) : (16 + kb + e - 8);
                        B[e] = lp[kk * 3 * ROWP];
                    }
                    acc[j] = __builtin_amdgcn_wmma_f32_16x16x32_f16(
                                 false, A[s], false, B, (short)0, acc[j], false, false);
                }
            }
        }
    }

    // epilogue: bias + lrelu*sqrt(2) + sigmoid; D layout: N=lane%16, M=v+8*(lane/16)
    const float sq2  = 1.41421356237309515f;
    const int   nn   = lane & 15;
    const int   mtop = (lane >> 4) * 8;
    const size_t obase = (((size_t)n * C_OUT + coBase) * HH + h) * WWID;
#pragma unroll
    for (int j = 0; j < 4; ++j) {
        const int px0 = pxHalf * 64 + j * 16;
#pragma unroll
        for (int v = 0; v < 8; ++v) {
            const int coOff = v + mtop;
            float o = acc[j][v] + bias[coBase + coOff];
            o = (o >= 0.f ? o : 0.2f * o) * sq2;
            o = 1.f / (1.f + __expf(-o));
            out[obase + (size_t)coOff * HH * WWID + px0 + nn] = o;
        }
    }
}

extern "C" void kernel_launch(void* const* d_in, const int* in_sizes, int n_in,
                              void* d_out, int out_size, void* d_ws, size_t ws_size,
                              hipStream_t stream) {
    (void)in_sizes; (void)n_in; (void)out_size; (void)ws_size;
    const float* x   = (const float*)d_in[0];
    const float* w   = (const float*)d_in[1];
    const float* aw  = (const float*)d_in[2];
    const float* ab  = (const float*)d_in[3];
    const float* wt  = (const float*)d_in[4];
    const float* bs  = (const float*)d_in[5];
    float* out = (float*)d_out;

    float*    styles = (float*)d_ws;                                 // 8 KB
    _Float16* wwt    = (_Float16*)((char*)d_ws + 8192);              // 9.4 MB

    k_styles<<<dim3(NB), dim3(256), 0, stream>>>(w, aw, ab, styles);
    k_modw  <<<dim3(C_OUT, NB), dim3(256), 0, stream>>>(wt, styles, wwt);
    k_conv  <<<dim3(4, HH, NB), dim3(256), 0, stream>>>(x, wwt, bs, out);
}
